// Decoder_5669356831107
// MI455X (gfx1250) — matrix operand order repaired and checked by reading, not compiled, not run on hardware
//
#include <hip/hip_runtime.h>
#include <hip/hip_bf16.h>
#include <math.h>

// Sizes from the reference
#define BB 64
#define PP 196
#define EE 2048
#define LL 52
#define VV 20000
#define AA 512
#define MM 512
#define HH 512
#define TT 51          // L-1 timesteps
#define XK 3072        // (M + E) + H : combined LSTM GEMM K
#define G4H 2048       // 4*H

typedef __attribute__((ext_vector_type(16))) __bf16 v16bf;
typedef __attribute__((ext_vector_type(8)))  float  v8f;

__device__ __forceinline__ unsigned short f2bf(float f) {
    unsigned int u = __float_as_uint(f);
    u += 0x7fffu + ((u >> 16) & 1u);       // round to nearest even
    return (unsigned short)(u >> 16);
}
__device__ __forceinline__ float bf2f(unsigned short h) {
    return __uint_as_float(((unsigned int)h) << 16);
}
__device__ __forceinline__ float sigm(float x) { return 1.0f / (1.0f + expf(-x)); }

// ---------------------------------------------------------------------------
// bf16 WMMA GEMM:  C[M x N] (f32) = A[M x K] (bf16 row-major)
//                                  @ WT[N x K] (bf16, i.e. B transposed) + bias
// One wave per 16(M) x 64(N) output macro-tile: 4 accumulators share one A
// fragment -> 4 WMMAs per K-slab.  Double-buffered K pipeline so next slab's
// loads overlap current slab's WMMAs.  K % 32 == 0, M % 16 == 0; N tiles
// beyond N/16 are clamped for loads and skipped on store (N=20000 case).
// Fragment k order is contiguous (k = 16*hi + s), same for A and B.
//
// MASKED epilogue (fc GEMM): row r is a batch row; write
//   Cout[row*ldc + coff + col] = (dec_len[row] > t) ? val : 0
// placing masked predictions directly in d_out (no temp round-trip).
// ---------------------------------------------------------------------------
union FragU { unsigned int u[8]; v16bf v; };

__device__ __forceinline__ void load_slab(const unsigned short* __restrict__ Arow,
                                          const unsigned short* const* Brow,
                                          int k0, int hi,
                                          FragU& a, FragU* b)
{
#pragma unroll
    for (int v = 0; v < 8; ++v) {
        const int ka = k0 + (hi << 4) + (v << 1);
        a.u[v] = *(const unsigned int*)(Arow + ka);
        const int kb = k0 + (hi << 4) + (v << 1);
#pragma unroll
        for (int j = 0; j < 4; ++j)
            b[j].u[v] = *(const unsigned int*)(Brow[j] + kb);
    }
}

template<bool MASKED>
__global__ __launch_bounds__(32)
void gemm_bf16_wmma(const unsigned short* __restrict__ A,
                    const unsigned short* __restrict__ WT,
                    const float* __restrict__ bias,
                    float* __restrict__ C,
                    int M, int N, int K,
                    const int* __restrict__ dec_len, int t,
                    size_t ldc, size_t coff)
{
    const int lane = threadIdx.x & 31;
    const int lo = lane & 15;
    const int hi = lane >> 4;
    const int mtile = blockIdx.y;
    const int ntiles = N >> 4;
    const int n0 = blockIdx.x * 4;

    const unsigned short* Arow = A + (size_t)(mtile * 16 + lo) * K;
    const unsigned short* Brow[4];
#pragma unroll
    for (int j = 0; j < 4; ++j) {
        int nt = n0 + j;
        if (nt >= ntiles) nt = ntiles - 1;     // clamp (uniform), skip on store
        Brow[j] = WT + (size_t)(nt * 16 + lo) * K;
    }

    v8f acc[4] = {};
    FragU a0, a1, b0[4], b1[4];
    load_slab(Arow, Brow, 0, hi, a0, b0);

    int k0 = 0;
    for (; k0 + 32 < K; k0 += 32) {
        load_slab(Arow, Brow, k0 + 32, hi, a1, b1);
        __builtin_prefetch(Arow + k0 + 64, 0, 3);
#pragma unroll
        for (int j = 0; j < 4; ++j)
            __builtin_prefetch(Brow[j] + k0 + 64, 0, 3);
#pragma unroll
        for (int j = 0; j < 4; ++j)
            acc[j] = __builtin_amdgcn_wmma_f32_16x16x32_bf16(
                         false, a0.v, false, b0[j].v, (short)0, acc[j], false, false);
        a0 = a1;
#pragma unroll
        for (int j = 0; j < 4; ++j) b0[j] = b1[j];
    }
#pragma unroll
    for (int j = 0; j < 4; ++j)
        acc[j] = __builtin_amdgcn_wmma_f32_16x16x32_bf16(
                     false, a0.v, false, b0[j].v, (short)0, acc[j], false, false);

    const int row0 = mtile * 16 + hi * 8;
#pragma unroll
    for (int j = 0; j < 4; ++j) {
        if (n0 + j >= ntiles) continue;
        const int col = (n0 + j) * 16 + lo;
        const float bz = bias ? bias[col] : 0.0f;
#pragma unroll
        for (int r = 0; r < 8; ++r) {
            const int row = row0 + r;
            float val = acc[j][r] + bz;
            if (MASKED) {
                const bool active = dec_len[row] > t;
                val = active ? val : 0.0f;
            }
            C[(size_t)row * ldc + coff + col] = val;
        }
    }
}

// ---------------------------------------------------------------------------
// Stable descending argsort of lengths (jnp.argsort(-lengths)); writes sorted
// caps / dec_len / sort_idx, both to workspace (int) and d_out (float).
// ---------------------------------------------------------------------------
__global__ void sort_kernel(const int* __restrict__ caption_lengths,
                            const int* __restrict__ encoded_captions,
                            int* __restrict__ sort_idx,
                            int* __restrict__ caps_s,
                            int* __restrict__ dec_len,
                            float* __restrict__ out_caps,
                            float* __restrict__ out_declen,
                            float* __restrict__ out_sidx)
{
    __shared__ int len[BB];
    const int b = threadIdx.x;
    len[b] = caption_lengths[b];
    __syncthreads();
    const int myLen = len[b];
    int rank = 0;
    for (int j = 0; j < BB; ++j) {
        const int lj = len[j];
        if (lj > myLen || (lj == myLen && j < b)) ++rank;
    }
    sort_idx[rank]  = b;
    dec_len[rank]   = myLen - 1;
    out_declen[rank] = (float)(myLen - 1);
    out_sidx[rank]   = (float)b;
    for (int t = 0; t < LL; ++t) {
        const int v = encoded_captions[b * LL + t];
        caps_s[rank * LL + t]  = v;
        out_caps[rank * LL + t] = (float)v;
    }
}

// Gather sorted encoder features and convert to bf16.
__global__ void gather_enc_kernel(const float* __restrict__ enc,
                                  const int* __restrict__ sidx,
                                  unsigned short* __restrict__ enc_bf)
{
    const size_t idx = (size_t)blockIdx.x * blockDim.x + threadIdx.x;
    const size_t total = (size_t)BB * PP * EE;
    if (idx >= total) return;
    const int r = (int)(idx / ((size_t)PP * EE));
    const size_t rem = idx % ((size_t)PP * EE);
    enc_bf[idx] = f2bf(enc[(size_t)sidx[r] * PP * EE + rem]);
}

// mean over P of sorted encoder -> bf16 (64 x 2048)
__global__ void mean_enc_kernel(const float* __restrict__ enc,
                                const int* __restrict__ sidx,
                                unsigned short* __restrict__ mean_bf)
{
    const int idx = blockIdx.x * blockDim.x + threadIdx.x;
    if (idx >= BB * EE) return;
    const int r = idx >> 11, e = idx & (EE - 1);
    const float* p = enc + (size_t)sidx[r] * PP * EE + e;
    float s = 0.0f;
    for (int q = 0; q < PP; ++q) s += p[(size_t)q * EE];
    mean_bf[idx] = f2bf(s * (1.0f / (float)PP));
}

// Transpose-convert weight W[K x N] f32 -> WT bf16 with WT[n*ldt + koff + k]
__global__ void transpose_bf16_kernel(const float* __restrict__ W,
                                      unsigned short* __restrict__ WT,
                                      int K, int N, int ldt, int koff)
{
    const size_t idx = (size_t)blockIdx.x * blockDim.x + threadIdx.x;
    if (idx >= (size_t)K * N) return;
    const int k = (int)(idx / N);
    const int n = (int)(idx % N);
    WT[(size_t)n * ldt + koff + k] = f2bf(W[idx]);
}

__global__ void add_bias_kernel(const float* __restrict__ a,
                                const float* __restrict__ b,
                                float* __restrict__ o, int n)
{
    const int i = blockIdx.x * blockDim.x + threadIdx.x;
    if (i < n) o[i] = a[i] + b[i];
}

__global__ void f32_to_bf16_kernel(const float* __restrict__ s,
                                   unsigned short* __restrict__ d, int n)
{
    const int i = blockIdx.x * blockDim.x + threadIdx.x;
    if (i < n) d[i] = f2bf(s[i]);
}

// ---------------------------------------------------------------------------
// Attention scores + softmax.  One block per batch row.
// e[p] = relu(att1[r,p,:] + att2[r,:]) . w  + b ; alpha = softmax_p(e)
// ---------------------------------------------------------------------------
__global__ void attention_kernel(const float* __restrict__ att1,
                                 const float* __restrict__ att2,
                                 const float* __restrict__ wfull,
                                 const float* __restrict__ bfull,
                                 const int* __restrict__ dec_len, int t,
                                 float* __restrict__ alpha,
                                 float* __restrict__ alphas_out)
{
    const int r = blockIdx.x;
    const int tid = threadIdx.x;              // 256 threads
    __shared__ float sh_a2[AA];
    __shared__ float sh_w[AA];
    __shared__ float sh_red[256];

    for (int i = tid; i < AA; i += 256) {
        sh_a2[i] = att2[r * AA + i];
        sh_w[i]  = wfull[i];
    }
    __syncthreads();

    float e = -3.4e38f;
    if (tid < PP) {
        const float* a1 = att1 + ((size_t)(r * PP + tid)) * AA;
        float s = bfull[0];
        for (int j = 0; j < AA; ++j) {
            const float v = a1[j] + sh_a2[j];
            s += (v > 0.0f ? v : 0.0f) * sh_w[j];
        }
        e = s;
    }
    sh_red[tid] = e;
    __syncthreads();
    for (int s = 128; s > 0; s >>= 1) {
        if (tid < s) sh_red[tid] = fmaxf(sh_red[tid], sh_red[tid + s]);
        __syncthreads();
    }
    const float mx = sh_red[0];
    __syncthreads();
    const float ex = (tid < PP) ? expf(e - mx) : 0.0f;
    sh_red[tid] = ex;
    __syncthreads();
    for (int s = 128; s > 0; s >>= 1) {
        if (tid < s) sh_red[tid] += sh_red[tid + s];
        __syncthreads();
    }
    const float inv = 1.0f / sh_red[0];
    if (tid < PP) {
        const float a = ex * inv;
        alpha[r * PP + tid] = a;
        const bool active = dec_len[r] > t;
        alphas_out[((size_t)(r * TT + t)) * PP + tid] = active ? a : 0.0f;
    }
}

// ---------------------------------------------------------------------------
// Build bf16 LSTM input x = [emb_t | sigmoid(gate_lin) * awe | h]  (64 x 3072)
// awe[r,e] = sum_p alpha[r,p] * enc[r,p,e]
// ---------------------------------------------------------------------------
__global__ void build_x_kernel(const float* __restrict__ alpha,
                               const unsigned short* __restrict__ enc_bf,
                               const float* __restrict__ gate_lin,
                               const float* __restrict__ emb_W,
                               const int* __restrict__ caps_s,
                               const unsigned short* __restrict__ h_bf,
                               unsigned short* __restrict__ x_bf, int t)
{
    const int r = blockIdx.y;
    const int col = blockIdx.x * 256 + threadIdx.x;   // 0..3071
    __shared__ float sa[PP];
    if (threadIdx.x < PP) sa[threadIdx.x] = alpha[r * PP + threadIdx.x];
    __syncthreads();

    unsigned short out;
    if (col < MM) {
        const int w = caps_s[r * LL + t];
        out = f2bf(emb_W[(size_t)w * MM + col]);
    } else if (col < MM + EE) {
        const int e = col - MM;
        const unsigned short* ep = enc_bf + (size_t)r * PP * EE + e;
        float s = 0.0f;
        for (int p = 0; p < PP; ++p) s += sa[p] * bf2f(ep[(size_t)p * EE]);
        const float gate = sigm(gate_lin[r * EE + e]);
        out = f2bf(gate * s);
    } else {
        out = h_bf[r * HH + (col - MM - EE)];
    }
    x_bf[(size_t)r * XK + col] = out;
}

// LSTM cell elementwise + masking; keeps f32 h/c state and bf16 copies.
__global__ void lstm_cell_kernel(const float* __restrict__ g,
                                 float* __restrict__ h, float* __restrict__ c,
                                 unsigned short* __restrict__ h_bf,
                                 unsigned short* __restrict__ hnew_bf,
                                 const int* __restrict__ dec_len, int t)
{
    const int idx = blockIdx.x * 256 + threadIdx.x;  // 64*512
    if (idx >= BB * HH) return;
    const int r = idx >> 9, j = idx & (HH - 1);
    const float* gr = g + (size_t)r * G4H;
    const float gi = sigm(gr[j]);
    const float gf = sigm(gr[HH + j]);
    const float gg = tanhf(gr[2 * HH + j]);
    const float go = sigm(gr[3 * HH + j]);
    const float cn = gf * c[idx] + gi * gg;
    const float hn = go * tanhf(cn);
    hnew_bf[idx] = f2bf(hn);
    const bool active = dec_len[r] > t;
    const float hm = active ? hn : h[idx];
    const float cm = active ? cn : c[idx];
    h[idx] = hm; c[idx] = cm;
    h_bf[idx] = f2bf(hm);
}

// ---------------------------------------------------------------------------
#define CEILDIV(a, b) (((a) + (b) - 1) / (b))

static inline void launch_gemm(const unsigned short* A, const unsigned short* WT,
                               const float* bias, float* C,
                               int M, int N, int K, hipStream_t s)
{
    dim3 grid(CEILDIV(N / 16, 4), M / 16);
    gemm_bf16_wmma<false><<<grid, dim3(32), 0, s>>>(A, WT, bias, C, M, N, K,
                                                    nullptr, 0, (size_t)N, 0);
}

static inline void launch_gemm_masked(const unsigned short* A, const unsigned short* WT,
                                      const float* bias, float* C,
                                      int M, int N, int K,
                                      const int* dlen, int t, size_t ldc, size_t coff,
                                      hipStream_t s)
{
    dim3 grid(CEILDIV(N / 16, 4), M / 16);
    gemm_bf16_wmma<true><<<grid, dim3(32), 0, s>>>(A, WT, bias, C, M, N, K,
                                                   dlen, t, ldc, coff);
}

extern "C" void kernel_launch(void* const* d_in, const int* in_sizes, int n_in,
                              void* d_out, int out_size, void* d_ws, size_t ws_size,
                              hipStream_t stream)
{
    // ---- inputs (setup_inputs order) ----
    const float* enc      = (const float*)d_in[0];   // (64,196,2048)
    const int*   caps     = (const int*)d_in[1];     // (64,52)
    const int*   caplen   = (const int*)d_in[2];     // (64,1)
    const float* emb_W    = (const float*)d_in[3];   // (20000,512)
    const float* attencW  = (const float*)d_in[4];   // (2048,512)
    const float* attencB  = (const float*)d_in[5];   // (512)
    const float* attdecW  = (const float*)d_in[6];   // (512,512)
    const float* attdecB  = (const float*)d_in[7];   // (512)
    const float* attfullW = (const float*)d_in[8];   // (512,1)
    const float* attfullB = (const float*)d_in[9];   // (1)
    const float* Wih      = (const float*)d_in[10];  // (2560,2048)
    const float* Whh      = (const float*)d_in[11];  // (512,2048)
    const float* bih      = (const float*)d_in[12];  // (2048)
    const float* bhh      = (const float*)d_in[13];  // (2048)
    const float* inithW   = (const float*)d_in[14];  // (2048,512)
    const float* inithB   = (const float*)d_in[15];
    const float* initcW   = (const float*)d_in[16];  // (2048,512)
    const float* initcB   = (const float*)d_in[17];
    const float* fbetaW   = (const float*)d_in[18];  // (512,2048)
    const float* fbetaB   = (const float*)d_in[19];  // (2048)
    const float* fcW      = (const float*)d_in[20];  // (512,20000)
    const float* fcB      = (const float*)d_in[21];  // (20000)

    // ---- d_out layout (all f32): preds | caps | dec_len | alphas | sort_idx
    float* out = (float*)d_out;
    const size_t CAPS_OFF  = (size_t)BB * TT * VV;            // 65,280,000
    const size_t DECL_OFF  = CAPS_OFF + (size_t)BB * LL;
    const size_t ALPH_OFF  = DECL_OFF + BB;
    const size_t SIDX_OFF  = ALPH_OFF + (size_t)BB * TT * PP;

    // ---- workspace carve-up (~122 MB, 256B aligned) ----
    char* wp = (char*)d_ws;
    auto alloc = [&](size_t bytes) -> void* {
        void* p = (void*)wp;
        wp += (bytes + 255) & ~(size_t)255;
        return p;
    };
    unsigned short* enc_bf  = (unsigned short*)alloc((size_t)BB * PP * EE * 2);
    float*          att1    = (float*)alloc((size_t)BB * PP * AA * 4);
    unsigned short* attencT = (unsigned short*)alloc((size_t)AA * EE * 2);   // 512 x 2048
    unsigned short* attdecT = (unsigned short*)alloc((size_t)AA * HH * 2);   // 512 x 512
    unsigned short* fbetaT  = (unsigned short*)alloc((size_t)EE * HH * 2);   // 2048 x 512
    unsigned short* lstmT   = (unsigned short*)alloc((size_t)G4H * XK * 2);  // 2048 x 3072
    unsigned short* fcT     = (unsigned short*)alloc((size_t)VV * HH * 2);   // 20000 x 512
    unsigned short* inithT  = (unsigned short*)alloc((size_t)HH * EE * 2);
    unsigned short* initcT  = (unsigned short*)alloc((size_t)HH * EE * 2);
    float*          lstm_b  = (float*)alloc((size_t)G4H * 4);
    int*            sidx    = (int*)alloc((size_t)BB * 4);
    int*            caps_s  = (int*)alloc((size_t)BB * LL * 4);
    int*            dlen    = (int*)alloc((size_t)BB * 4);
    unsigned short* mean_bf = (unsigned short*)alloc((size_t)BB * EE * 2);
    float*          hbuf    = (float*)alloc((size_t)BB * HH * 4);
    float*          cbuf    = (float*)alloc((size_t)BB * HH * 4);
    unsigned short* h_bf    = (unsigned short*)alloc((size_t)BB * HH * 2);
    unsigned short* hnew_bf = (unsigned short*)alloc((size_t)BB * HH * 2);
    float*          att2    = (float*)alloc((size_t)BB * AA * 4);
    float*          alpha   = (float*)alloc((size_t)BB * PP * 4);
    float*          gatel   = (float*)alloc((size_t)BB * EE * 4);
    unsigned short* x_bf    = (unsigned short*)alloc((size_t)BB * XK * 2);
    float*          gbuf    = (float*)alloc((size_t)BB * G4H * 4);
    (void)ws_size; (void)out_size; (void)n_in; (void)in_sizes;

    // ---- one-time setup ----
    sort_kernel<<<1, BB, 0, stream>>>(caplen, caps, sidx, caps_s, dlen,
                                      out + CAPS_OFF, out + DECL_OFF, out + SIDX_OFF);
    {
        const size_t tot = (size_t)BB * PP * EE;
        gather_enc_kernel<<<(unsigned)CEILDIV(tot, 256), 256, 0, stream>>>(enc, sidx, enc_bf);
    }
    mean_enc_kernel<<<CEILDIV(BB * EE, 256), 256, 0, stream>>>(enc, sidx, mean_bf);

    // transpose-convert weights to bf16 (N x K)
    transpose_bf16_kernel<<<CEILDIV(EE * AA, 256), 256, 0, stream>>>(attencW, attencT, EE, AA, EE, 0);
    transpose_bf16_kernel<<<CEILDIV(HH * AA, 256), 256, 0, stream>>>(attdecW, attdecT, HH, AA, HH, 0);
    transpose_bf16_kernel<<<CEILDIV(HH * EE, 256), 256, 0, stream>>>(fbetaW, fbetaT, HH, EE, HH, 0);
    transpose_bf16_kernel<<<CEILDIV((MM + EE) * G4H, 256), 256, 0, stream>>>(Wih, lstmT, MM + EE, G4H, XK, 0);
    transpose_bf16_kernel<<<CEILDIV(HH * G4H, 256), 256, 0, stream>>>(Whh, lstmT, HH, G4H, XK, MM + EE);
    transpose_bf16_kernel<<<CEILDIV(HH * VV, 256), 256, 0, stream>>>(fcW, fcT, HH, VV, HH, 0);
    transpose_bf16_kernel<<<CEILDIV(EE * HH, 256), 256, 0, stream>>>(inithW, inithT, EE, HH, EE, 0);
    transpose_bf16_kernel<<<CEILDIV(EE * HH, 256), 256, 0, stream>>>(initcW, initcT, EE, HH, EE, 0);
    add_bias_kernel<<<CEILDIV(G4H, 256), 256, 0, stream>>>(bih, bhh, lstm_b, G4H);

    // h0 / c0 :  mean_enc @ init_{h,c}_W + b    (WMMA, 64x512, K=2048)
    launch_gemm(mean_bf, inithT, inithB, hbuf, BB, HH, EE, stream);
    launch_gemm(mean_bf, initcT, initcB, cbuf, BB, HH, EE, stream);
    f32_to_bf16_kernel<<<CEILDIV(BB * HH, 256), 256, 0, stream>>>(hbuf, h_bf, BB * HH);

    // att1 = enc_s @ att_enc_W + b   (WMMA, 12544 x 512, K=2048)
    launch_gemm(enc_bf, attencT, attencB, att1, BB * PP, AA, EE, stream);

    // ---- sequential decode: 51 steps ----
    for (int t = 0; t < TT; ++t) {
        // att2 = h @ att_dec_W + b         (64 x 512, K=512)
        launch_gemm(h_bf, attdecT, attdecB, att2, BB, AA, HH, stream);
        // gate_lin = h @ f_beta_W + b      (64 x 2048, K=512)
        launch_gemm(h_bf, fbetaT, fbetaB, gatel, BB, EE, HH, stream);
        // e / softmax / alpha (+ masked alphas output)
        attention_kernel<<<BB, 256, 0, stream>>>(att1, att2, attfullW, attfullB,
                                                 dlen, t, alpha, out + ALPH_OFF);
        // x = [emb | gate*awe | h]  (bf16, 64 x 3072)
        build_x_kernel<<<dim3(XK / 256, BB), 256, 0, stream>>>(alpha, enc_bf, gatel,
                                                               emb_W, caps_s, h_bf, x_bf, t);
        // g = x @ [Wih;Whh] + (bih+bhh)    (64 x 2048, K=3072)
        launch_gemm(x_bf, lstmT, lstm_b, gbuf, BB, G4H, XK, stream);
        // LSTM cell + masking
        lstm_cell_kernel<<<CEILDIV(BB * HH, 256), 256, 0, stream>>>(gbuf, hbuf, cbuf,
                                                                    h_bf, hnew_bf, dlen, t);
        // preds = h_new @ fc_W + fc_b, masked + placed directly into d_out:
        //   out[(r*TT + t)*VV + n]   (64 x 20000, K=512)
        launch_gemm_masked(hnew_bf, fcT, fcB, out, BB, VV, HH,
                           dlen, t, (size_t)TT * VV, (size_t)t * VV, stream);
    }
}
